// GatedDeltaNet_77841987273018
// MI455X (gfx1250) — matrix operand order repaired and checked by reading, not compiled, not run on hardware
//
#include <hip/hip_runtime.h>
#include <hip/hip_bf16.h>
#include <math.h>

// ---------------------------------------------------------------------------
// Gated DeltaNet forward for MI455X (gfx1250, wave32, WMMA).
//   1) fp32 -> bf16: x row-major; ALL weights transposed to [N,K] (B operands)
//   2) bf16 WMMA GEMMs (f32 accum), async global->LDS double-buffered tiles
//   3) causal depthwise conv (KC=4) + silu + per-head l2norm + gates (fp32)
//   4) sequential delta-rule scan, state 128x128 in VGPRs, chunked staging
//   5) gated RMSNorm -> bf16
//   6) bf16 WMMA GEMM: y @ Wout -> fp32 output
// ---------------------------------------------------------------------------

typedef __attribute__((ext_vector_type(16))) __bf16 v16bf;
typedef __attribute__((ext_vector_type(8)))  float  v8f;

#define DIMX 2048
#define HK 16
#define HV 32
#define DK 128
#define DV 128
#define KC 4
#define LL 8192
#define CONV_DIM (2*HK*DK + HV*DV)   // 8192
#define EPSF 1e-6f

#define BM 128
#define BN 128
#define BK 32

// ---- CDNA5 async global->LDS path (guarded; falls back to sync loads) ----
#if defined(__has_builtin)
#  if __has_builtin(__builtin_amdgcn_global_load_async_to_lds_b128) && \
      __has_builtin(__builtin_amdgcn_s_wait_asynccnt)
#    define USE_ASYNC_LDS 1
#  endif
#endif
#ifndef USE_ASYNC_LDS
#  define USE_ASYNC_LDS 0
#endif

#if USE_ASYNC_LDS
typedef int v4i_vs __attribute__((vector_size(16)));
typedef __attribute__((address_space(1))) v4i_vs* as1_v4i;
typedef __attribute__((address_space(3))) v4i_vs* as3_v4i;

static __device__ __forceinline__ void async_copy16(const unsigned short* g,
                                                    unsigned short* l) {
  // generic->AS1 / generic->AS3 via integer round-trip (LDS offset = low 32b)
  as1_v4i gp = (as1_v4i)(unsigned long long)g;
  as3_v4i lp = (as3_v4i)(unsigned int)(unsigned long long)l;
  __builtin_amdgcn_global_load_async_to_lds_b128(gp, lp, 0, 0);
}
#endif

static __device__ __forceinline__ unsigned short f32_to_bf16(float f) {
  unsigned int u = __float_as_uint(f);
  if ((u & 0x7fffffffu) > 0x7f800000u) return (unsigned short)((u >> 16) | 0x0040u);
  u += 0x7fffu + ((u >> 16) & 1u);   // round to nearest even
  return (unsigned short)(u >> 16);
}

// ---------------------------------------------------------------------------
// fp32 -> bf16 (row-major copy, for A operands)
// ---------------------------------------------------------------------------
__global__ void k_f32_to_bf16(const float* __restrict__ src,
                              unsigned short* __restrict__ dst, long n) {
  long i = (long)blockIdx.x * blockDim.x + threadIdx.x;
  long stride = (long)gridDim.x * blockDim.x;
  for (; i < n; i += stride) dst[i] = f32_to_bf16(src[i]);
}

// fp32 W[K,N] -> bf16 WT[n,k] (B operands; grid: (K/256, Npad))
__global__ void k_transpose_bf16(const float* __restrict__ src,
                                 unsigned short* __restrict__ dst,
                                 int K, int N) {
  int k = blockIdx.x * blockDim.x + threadIdx.x;
  int n = blockIdx.y;
  if (k >= K) return;
  float v = (n < N) ? src[(long)k * N + n] : 0.f;
  dst[(long)n * K + k] = f32_to_bf16(v);
}

// Wa|Wb -> bf16 [128, DIMX] transposed, rows 64..127 zero (grid: (DIMX/256,128))
__global__ void k_pack_ab_t_bf16(const float* __restrict__ Wa,
                                 const float* __restrict__ Wb,
                                 unsigned short* __restrict__ dst) {
  int k = blockIdx.x * blockDim.x + threadIdx.x;
  int n = blockIdx.y;
  if (k >= DIMX) return;
  float v = 0.f;
  if (n < HV)            v = Wa[(long)k * HV + n];
  else if (n < 2 * HV)   v = Wb[(long)k * HV + (n - HV)];
  dst[(long)n * DIMX + k] = f32_to_bf16(v);
}

// ---------------------------------------------------------------------------
// bf16 GEMM: C[M,N] = A[M,K] @ BT[N,K]^T.  A row-major bf16, BT row-major bf16.
// Requires: M % 128 == 0, K % 32 == 0, BT padded to >= ceil(N/128)*128 rows.
// 256 threads = 8 waves; block tile 128x128x32; wave tile 64x32 (4x2 WMMAs).
// Double-buffered LDS tiles fed by async global->LDS (ASYNCcnt) when available.
// ---------------------------------------------------------------------------
__global__ __launch_bounds__(256)
void k_gemm_bf16(const unsigned short* __restrict__ A,
                 const unsigned short* __restrict__ BT,
                 float* __restrict__ C, int M, int N, int K) {
  __shared__ __align__(16) unsigned short lA[2][BM * BK];   // [m][k]
  __shared__ __align__(16) unsigned short lB[2][BN * BK];   // [n][k]

  const int tid  = threadIdx.x;
  const int lane = tid & 31;
  const int wave = tid >> 5;
  const int wm   = wave >> 2;          // 0..1 (64-row slab)
  const int wn   = wave & 3;           // 0..3 (32-col slab)
  const int tileM = blockIdx.y * BM;
  const int tileN = blockIdx.x * BN;

  const int row16 = tid >> 1;          // 0..127
  const int col16 = (tid & 1) << 4;    // 0 or 16

  const unsigned short* gA = A  + (long)(tileM + row16) * K + col16;
  const unsigned short* gB = BT + (long)(tileN + row16) * K + col16;
  unsigned short* sA = &lA[0][row16 * BK + col16];
  unsigned short* sB = &lB[0][row16 * BK + col16];

  v8f acc[4][2];
  v8f zero = {};
  #pragma unroll
  for (int i = 0; i < 4; ++i)
    #pragma unroll
    for (int j = 0; j < 2; ++j) acc[i][j] = zero;

  auto stage = [&](int buf, int k0) {
#if USE_ASYNC_LDS
    async_copy16(gA + k0,     sA + buf * (BM * BK));
    async_copy16(gA + k0 + 8, sA + buf * (BM * BK) + 8);
    async_copy16(gB + k0,     sB + buf * (BN * BK));
    async_copy16(gB + k0 + 8, sB + buf * (BN * BK) + 8);
#else
    uint4 a0 = *(const uint4*)(gA + k0);
    uint4 a1 = *(const uint4*)(gA + k0 + 8);
    uint4 b0 = *(const uint4*)(gB + k0);
    uint4 b1 = *(const uint4*)(gB + k0 + 8);
    *(uint4*)(sA + buf * (BM * BK))     = a0;
    *(uint4*)(sA + buf * (BM * BK) + 8) = a1;
    *(uint4*)(sB + buf * (BN * BK))     = b0;
    *(uint4*)(sB + buf * (BN * BK) + 8) = b1;
    if (k0 + BK < K) __builtin_prefetch(gA + k0 + BK, 0, 1);
#endif
  };

  const int nk = K / BK;
  stage(0, 0);

  for (int ks = 0; ks < nk; ++ks) {
#if USE_ASYNC_LDS
    __builtin_amdgcn_s_wait_asynccnt(0);
#endif
    __syncthreads();
    if (ks + 1 < nk) stage((ks + 1) & 1, (ks + 1) * BK);

    const unsigned short* cA = &lA[ks & 1][0];
    const unsigned short* cB = &lB[ks & 1][0];
    const int half = lane >> 4;
    const int m16  = lane & 15;

    v16bf afrag[4];
    #pragma unroll
    for (int i = 0; i < 4; ++i) {
      int r = wm * 64 + i * 16 + m16;
      uint4* fp = (uint4*)&afrag[i];
      fp[0] = *(const uint4*)&cA[r * BK + half * 8];        // K = half*8 + 0..7
      fp[1] = *(const uint4*)&cA[r * BK + 16 + half * 8];   // K = 16+half*8 ..
    }
    v16bf bfrag[2];
    #pragma unroll
    for (int j = 0; j < 2; ++j) {
      int n = wn * 32 + j * 16 + m16;
      uint4* fp = (uint4*)&bfrag[j];
      fp[0] = *(const uint4*)&cB[n * BK + half * 8];        // K = half*8 + 0..7
      fp[1] = *(const uint4*)&cB[n * BK + 16 + half * 8];   // K = 16+half*8 ..
    }
    #pragma unroll
    for (int i = 0; i < 4; ++i)
      #pragma unroll
      for (int j = 0; j < 2; ++j)
        acc[i][j] = __builtin_amdgcn_wmma_f32_16x16x32_bf16(
            false, afrag[i], false, bfrag[j], (short)0, acc[i][j],
            false, false);
    __syncthreads();
  }

  // epilogue: C/D layout lane<16 -> M=r, lane>=16 -> M=8+r; N=lane&15
  const int n16 = lane & 15;
  const int hi  = lane >> 4;
  #pragma unroll
  for (int i = 0; i < 4; ++i)
    #pragma unroll
    for (int j = 0; j < 2; ++j)
      #pragma unroll
      for (int r = 0; r < 8; ++r) {
        int row = tileM + wm * 64 + i * 16 + (hi ? 8 + r : r);
        int col = tileN + wn * 32 + j * 16 + n16;
        if (row < M && col < N) C[(long)row * N + col] = acc[i][j][r];
      }
}

// ---------------------------------------------------------------------------
// Causal depthwise conv + silu + per-head l2norm + gates. One block per l.
// ---------------------------------------------------------------------------
__global__ __launch_bounds__(256)
void k_conv_gates(const float* __restrict__ qkv, const float* __restrict__ ab,
                  const float* __restrict__ conv_w, const float* __restrict__ A_log,
                  const float* __restrict__ dt_bias, const long long* __restrict__ cu,
                  int nseq,
                  float* __restrict__ qn, float* __restrict__ kn,
                  float* __restrict__ vv, float* __restrict__ g,
                  float* __restrict__ beta) {
  __shared__ float buf[CONV_DIM];      // 32 KB
  __shared__ float nrm[2 * HK];
  const int l   = blockIdx.x;
  const int tid = threadIdx.x;

  long start = 0;
  for (int s = 0; s < nseq; ++s)
    if ((long)l >= (long)cu[s] && (long)l < (long)cu[s + 1]) { start = (long)cu[s]; break; }

  for (int c = tid; c < CONV_DIM; c += 256) {
    float acc = 0.f;
    #pragma unroll
    for (int i = 0; i < KC; ++i) {
      long lp = (long)l + i - (KC - 1);
      if (lp >= start) acc += conv_w[c * KC + i] * qkv[lp * CONV_DIM + c];
    }
    buf[c] = acc / (1.f + __expf(-acc));         // silu
  }
  __syncthreads();

  const int lane = tid & 31;
  const int wave = tid >> 5;
  for (int grp = wave; grp < 2 * HK; grp += 8) {
    float s = 0.f;
    #pragma unroll
    for (int e = 0; e < 4; ++e) {
      float x = buf[grp * DK + lane * 4 + e];
      s += x * x;
    }
    #pragma unroll
    for (int off = 16; off > 0; off >>= 1) s += __shfl_xor(s, off, 32);
    if (lane == 0) nrm[grp] = rsqrtf(s + EPSF);
  }
  __syncthreads();

  const float qscale = 0.08838834764831845f;     // DK^-0.5
  for (int idx = tid; idx < HK * DK; idx += 256) {
    int h = idx >> 7;
    qn[(long)l * (HK * DK) + idx] = buf[idx] * nrm[h] * qscale;
    kn[(long)l * (HK * DK) + idx] = buf[HK * DK + idx] * nrm[HK + h];
  }
  for (int idx = tid; idx < HV * DV; idx += 256)
    vv[(long)l * (HV * DV) + idx] = buf[2 * HK * DK + idx];

  if (tid < HV) {
    float av = ab[(long)l * 64 + tid] + dt_bias[tid];
    float sp = (av > 20.f) ? av : log1pf(__expf(av));
    g[(long)l * HV + tid] = -__expf(A_log[tid]) * sp;
    float bv = ab[(long)l * 64 + HV + tid];
    beta[(long)l * HV + tid] = 1.f / (1.f + __expf(-bv));
  }
}

// ---------------------------------------------------------------------------
// Sequential delta-rule scan. One block per (head, sequence); 128 threads.
// Lane t owns state column S[:, t] in VGPRs. q/k/v/g/beta staged 8 steps/chunk.
// ---------------------------------------------------------------------------
#define SCH 8
__global__ __launch_bounds__(128)
void k_scan(const float* __restrict__ qn, const float* __restrict__ kn,
            const float* __restrict__ vv, const float* __restrict__ g,
            const float* __restrict__ beta, const long long* __restrict__ cu,
            float* __restrict__ o) {
  const int hv   = blockIdx.x & (HV - 1);
  const int sidx = blockIdx.x >> 5;
  const int hk   = hv >> 1;            // rep = HV/HK = 2
  const int tid  = threadIdx.x;
  __shared__ float lq[SCH][DK], lk[SCH][DK], lv[SCH][DV];
  __shared__ float sgb[SCH][2];

  const long start = (long)cu[sidx];
  const long end   = (long)cu[sidx + 1];

  float S[DK];
  #pragma unroll
  for (int i = 0; i < DK; ++i) S[i] = 0.f;

  for (long l0 = start; l0 < end; l0 += SCH) {
    long rem = end - l0;
    int cn = rem < SCH ? (int)rem : SCH;
    for (int t = 0; t < cn; ++t) {
      long l = l0 + t;
      lq[t][tid] = qn[(l * HK + hk) * DK + tid];
      lk[t][tid] = kn[(l * HK + hk) * DK + tid];
      lv[t][tid] = vv[(l * HV + hv) * DV + tid];
    }
    if (tid < cn) {
      sgb[tid][0] = __expf(g[(l0 + tid) * HV + hv]);
      sgb[tid][1] = beta[(l0 + tid) * HV + hv];
    }
    __syncthreads();
    for (int t = 0; t < cn; ++t) {
      const float dec = sgb[t][0], bt = sgb[t][1];
      float kv = 0.f;
      #pragma unroll
      for (int kk = 0; kk < DK; ++kk) { S[kk] *= dec; kv += lk[t][kk] * S[kk]; }
      const float delta = (lv[t][tid] - kv) * bt;
      float oacc = 0.f;
      #pragma unroll
      for (int kk = 0; kk < DK; ++kk) { S[kk] += lk[t][kk] * delta; oacc += lq[t][kk] * S[kk]; }
      o[((l0 + t) * HV + hv) * DV + tid] = oacc;
    }
    __syncthreads();
  }
}

// ---------------------------------------------------------------------------
// Gated RMSNorm: y = o * silu(z); y *= rsqrt(mean(y^2)+eps) * w; -> bf16
// ---------------------------------------------------------------------------
__global__ __launch_bounds__(128)
void k_gated_rmsnorm(const float* __restrict__ o, const float* __restrict__ z,
                     const float* __restrict__ norm_w,
                     unsigned short* __restrict__ ybf) {
  const int row = blockIdx.x;          // l*HV + hv
  const int l   = row >> 5;
  const int hv  = row & 31;
  const int tid = threadIdx.x;
  __shared__ float wsum[4];

  float ov = o[(long)row * DV + tid];
  float zv = z[(long)l * (HV * DV) + hv * DV + tid];
  float y  = ov * (zv / (1.f + __expf(-zv)));
  float ss = y * y;
  #pragma unroll
  for (int off = 16; off > 0; off >>= 1) ss += __shfl_xor(ss, off, 32);
  if ((tid & 31) == 0) wsum[tid >> 5] = ss;
  __syncthreads();
  float tot = wsum[0] + wsum[1] + wsum[2] + wsum[3];
  y = y * rsqrtf(tot * (1.f / DV) + EPSF) * norm_w[tid];
  ybf[(long)l * (HV * DV) + hv * DV + tid] = f32_to_bf16(y);
}

// ---------------------------------------------------------------------------
// Host-side launcher
// ---------------------------------------------------------------------------
extern "C" void kernel_launch(void* const* d_in, const int* in_sizes, int n_in,
                              void* d_out, int out_size, void* d_ws, size_t ws_size,
                              hipStream_t stream) {
  const float*     x       = (const float*)d_in[0];
  const long long* cu      = (const long long*)d_in[1];
  const float*     Wqkv    = (const float*)d_in[2];
  const float*     Wz      = (const float*)d_in[3];
  const float*     Wa      = (const float*)d_in[4];
  const float*     Wb      = (const float*)d_in[5];
  const float*     conv_w  = (const float*)d_in[6];
  const float*     A_log   = (const float*)d_in[7];
  const float*     dt_bias = (const float*)d_in[8];
  const float*     norm_w  = (const float*)d_in[9];
  const float*     Wout    = (const float*)d_in[10];
  float* out = (float*)d_out;
  const int nseq = in_sizes[1] - 1;

  char* ws = (char*)d_ws;
  size_t off = 0;
  auto alloc = [&](size_t bytes) -> size_t {
    size_t o = off;
    off += (bytes + 255) & ~(size_t)255;
    return o;
  };

  unsigned short* xbf    = (unsigned short*)(ws + alloc((size_t)LL * DIMX * 2));
  unsigned short* wqkvbf = (unsigned short*)(ws + alloc((size_t)CONV_DIM * DIMX * 2)); // [N,K]
  unsigned short* wzbf   = (unsigned short*)(ws + alloc((size_t)HV * DV * DIMX * 2));  // [N,K]
  unsigned short* wabbf  = (unsigned short*)(ws + alloc((size_t)128 * DIMX * 2));      // [128,K] padded
  unsigned short* woutbf = (unsigned short*)(ws + alloc((size_t)DIMX * HV * DV * 2));  // [N,K]
  float* qkv  = (float*)(ws + alloc((size_t)LL * CONV_DIM * 4));   // reused as o
  float* zbuf = (float*)(ws + alloc((size_t)LL * HV * DV * 4));
  float* ab   = (float*)(ws + alloc((size_t)LL * 64 * 4));
  float* qn   = (float*)(ws + alloc((size_t)LL * HK * DK * 4));    // reused as ybf
  float* kn   = (float*)(ws + alloc((size_t)LL * HK * DK * 4));
  float* vv   = (float*)(ws + alloc((size_t)LL * HV * DV * 4));
  float* gbuf = (float*)(ws + alloc((size_t)LL * HV * 4));
  float* bbuf = (float*)(ws + alloc((size_t)LL * HV * 4));
  float* obuf = qkv;                         // qkv fully consumed before scan
  unsigned short* ybf = (unsigned short*)qn; // qn fully consumed before norm

  // 1) conversions (A row-major; weights transposed to [N,K])
  k_f32_to_bf16<<<dim3(2048), 256, 0, stream>>>(x, xbf, (long)LL * DIMX);
  k_transpose_bf16<<<dim3(DIMX / 256, CONV_DIM), 256, 0, stream>>>(Wqkv, wqkvbf, DIMX, CONV_DIM);
  k_transpose_bf16<<<dim3(DIMX / 256, HV * DV), 256, 0, stream>>>(Wz, wzbf, DIMX, HV * DV);
  k_pack_ab_t_bf16<<<dim3(DIMX / 256, 128), 256, 0, stream>>>(Wa, Wb, wabbf);
  k_transpose_bf16<<<dim3((HV * DV) / 256, DIMX), 256, 0, stream>>>(Wout, woutbf, HV * DV, DIMX);

  // 2) projections (WMMA bf16, f32 accumulate)
  k_gemm_bf16<<<dim3(CONV_DIM / BN, LL / BM), 256, 0, stream>>>(
      xbf, wqkvbf, qkv, LL, CONV_DIM, DIMX);
  k_gemm_bf16<<<dim3(HV * DV / BN, LL / BM), 256, 0, stream>>>(
      xbf, wzbf, zbuf, LL, HV * DV, DIMX);
  k_gemm_bf16<<<dim3(1, LL / BM), 256, 0, stream>>>(
      xbf, wabbf, ab, LL, 64, DIMX);

  // 3) conv + silu + l2norm + gates
  k_conv_gates<<<dim3(LL), 256, 0, stream>>>(qkv, ab, conv_w, A_log, dt_bias,
                                             cu, nseq, qn, kn, vv, gbuf, bbuf);

  // 4) recurrent scan
  k_scan<<<dim3(HV * nseq), 128, 0, stream>>>(qn, kn, vv, gbuf, bbuf, cu, obuf);

  // 5) gated RMSNorm -> bf16
  k_gated_rmsnorm<<<dim3(LL * HV), 128, 0, stream>>>(obuf, zbuf, norm_w, ybf);

  // 6) output projection (WMMA)
  k_gemm_bf16<<<dim3(DIMX / BN, LL / BM), 256, 0, stream>>>(
      ybf, woutbf, out, LL, DIMX, HV * DV);
}
